// MultiHeadAttention_36206574305617
// MI455X (gfx1250) — matrix operand order repaired and checked by reading, not compiled, not run on hardware
//
#include <hip/hip_runtime.h>
#include <hip/hip_bf16.h>

#define MODEL_DIM 1024
#define NUM_HEADS 16
#define D_K 64
#define S_LEN 1024
#define BATCH 8

typedef __attribute__((ext_vector_type(16))) __bf16 v16bf;
typedef __attribute__((ext_vector_type(8)))  __bf16 v8bf;
typedef __attribute__((ext_vector_type(8)))  float  v8f;

__device__ __forceinline__ v8f wmma_bf16(v16bf a, v16bf b, v8f c) {
    return __builtin_amdgcn_wmma_f32_16x16x32_bf16(false, a, false, b, (short)0, c, false, false);
}

// CDNA5 async global->LDS DMA (tracked by ASYNCcnt). VDST operand carries the
// LDS byte address; a generic pointer to __shared__ has that offset in its
// low 32 bits (LDS aperture base lives in addr[63:32]).
__device__ __forceinline__ void async_cp16(const __bf16* g, const __bf16* l) {
    const unsigned lds = (unsigned)(uintptr_t)l;
    asm volatile("global_load_async_to_lds_b128 %0, %1, off"
                 :: "v"(lds), "v"(g) : "memory");
}

// ---------------------------------------------------------------------------
// Elementwise f32 -> bf16 (one-time; removes conversions from GEMM hot loops)
// ---------------------------------------------------------------------------
__global__ __launch_bounds__(256) void cvt_bf16(const float* __restrict__ in,
                                                __bf16* __restrict__ out, int n) {
    const int i = (blockIdx.x * 256 + threadIdx.x) * 8;
    if (i >= n) return;
    const float4* p = (const float4*)(in + i);
    const float4 x0 = p[0], x1 = p[1];
    v8bf o;
    o[0] = (__bf16)x0.x; o[1] = (__bf16)x0.y; o[2] = (__bf16)x0.z; o[3] = (__bf16)x0.w;
    o[4] = (__bf16)x1.x; o[5] = (__bf16)x1.y; o[6] = (__bf16)x1.z; o[7] = (__bf16)x1.w;
    *(v8bf*)(out + i) = o;
}

// ---------------------------------------------------------------------------
// Wave-tiled GEMM (all-bf16 operands): out[m,n] = sum_k X[m,k]*W[n,k] + bias[n]
// Each wave computes a 32x64 tile: 2 A-fragments x 4 B-fragments = 8 WMMA/k-step.
// mode 0: f32 out [M, MODEL_DIM]
// mode 1: bf16 out [B,H,S,dk]   (Q/K head layout)
// mode 2: bf16 out [B,H,dk,S]   (V transposed so PV B-fragments are contiguous)
// ---------------------------------------------------------------------------
__global__ __launch_bounds__(128) void proj_wmma(
    const __bf16* __restrict__ X, const __bf16* __restrict__ W,
    const float* __restrict__ bias, float* __restrict__ outF,
    __bf16* __restrict__ outB, int mode)
{
    const int Kd   = MODEL_DIM;
    const int lane = threadIdx.x & 31;
    const int wave = threadIdx.x >> 5;
    const int gw   = blockIdx.x * 4 + wave;
    const int nBlk = MODEL_DIM / 64;
    const int row0 = (gw / nBlk) * 32;
    const int col0 = (gw % nBlk) * 64;
    const int half = lane >> 4;
    const int l15  = lane & 15;
    const int kbA  = half * 16;
    const int kbB  = half * 16;

    v8f acc[2][4];
#pragma unroll
    for (int i = 0; i < 2; ++i)
#pragma unroll
        for (int t = 0; t < 4; ++t)
#pragma unroll
            for (int r = 0; r < 8; ++r) acc[i][t][r] = 0.0f;

    for (int k0 = 0; k0 < Kd; k0 += 32) {
        v16bf a[2];
#pragma unroll
        for (int i = 0; i < 2; ++i) {
            const __bf16* xr = X + (size_t)(row0 + i * 16 + l15) * Kd + k0 + kbA;
            const v8bf lo = *(const v8bf*)xr;
            const v8bf hi = *(const v8bf*)(xr + 8);
#pragma unroll
            for (int e = 0; e < 8; ++e) { a[i][e] = lo[e]; a[i][8 + e] = hi[e]; }
        }
#pragma unroll
        for (int t = 0; t < 4; ++t) {
            const int n = col0 + t * 16 + l15;
            const v16bf bf = *(const v16bf*)(W + (size_t)n * Kd + k0 + kbB);
            acc[0][t] = wmma_bf16(a[0], bf, acc[0][t]);
            acc[1][t] = wmma_bf16(a[1], bf, acc[1][t]);
        }
    }

#pragma unroll
    for (int t = 0; t < 4; ++t) {
        const int n = col0 + t * 16 + l15;
        const float bv = bias[n];
#pragma unroll
        for (int i = 0; i < 2; ++i) {
#pragma unroll
            for (int r = 0; r < 8; ++r) {
                const int mr = row0 + i * 16 + r + half * 8;
                const float val = acc[i][t][r] + bv;
                if (mode == 0) {
                    outF[(size_t)mr * MODEL_DIM + n] = val;
                } else {
                    const int b = mr / S_LEN, s = mr % S_LEN;
                    const int h = n / D_K,   d = n % D_K;
                    if (mode == 1)
                        outB[(((size_t)(b * NUM_HEADS + h)) * S_LEN + s) * D_K + d] = (__bf16)val;
                    else
                        outB[(((size_t)(b * NUM_HEADS + h)) * D_K + d) * S_LEN + s] = (__bf16)val;
                }
            }
        }
    }
}

// ---------------------------------------------------------------------------
// Fused flash attention. WG = 256 threads (8 waves) = 128 query rows of one
// (b,h). K/V streamed in 64-wide chunks, staged in LDS once per WG via
// double-buffered GLOBAL_LOAD_ASYNC_TO_LDS_B128 (overlaps DMA with WMMA),
// cutting redundant global K/V traffic 8x versus per-wave loading.
// Q,K: bf16 [B,H,S,dk]; Vt: bf16 [B,H,dk,S]; Cout: bf16 [B,S,H*dk]
// ---------------------------------------------------------------------------
__global__ __launch_bounds__(256) void attn_wmma(
    const __bf16* __restrict__ Q, const __bf16* __restrict__ Kb,
    const __bf16* __restrict__ Vt, const int* __restrict__ mask,
    __bf16* __restrict__ Cout)
{
    // 72 bf16 = 144 B row stride: 16B-aligned pieces, 36-dword stride ->
    // conflict-free across a 16-lane fragment read.
    __shared__ __align__(16) __bf16 kbuf[2][64][72];
    __shared__ __align__(16) __bf16 vbuf[2][64][72];
    __shared__ __bf16 pbuf[8][16][66];

    const int tid  = threadIdx.x;
    const int lane = tid & 31;
    const int wave = tid >> 5;
    const int l15  = lane & 15;
    const int half = lane >> 4;

    const int qblocks = S_LEN / 128;               // 8 q-segments per (b,h)
    const int bh   = blockIdx.x / qblocks;
    const int qblk = blockIdx.x % qblocks;
    const int b    = bh / NUM_HEADS;
    const int h    = bh % NUM_HEADS;
    const int q0   = qblk * 128 + wave * 16;

    const __bf16* Qh = Q  + (size_t)bh * S_LEN * D_K;
    const __bf16* Kh = Kb + (size_t)bh * S_LEN * D_K;
    const __bf16* Vh = Vt + (size_t)bh * D_K * S_LEN;
    const int*  mrow = mask + (size_t)b * S_LEN;

    const int kbA = half * 16;
    const int kbB = half * 16;

    // Stage one 64-wide K/V chunk: 512 16B pieces each, 4 async ops/thread.
    auto stage = [&](int c, int buf) {
        const int kv0 = c * 64;
#pragma unroll
        for (int rep = 0; rep < 2; ++rep) {
            const int p   = tid + rep * 256;       // 0..511
            const int row = p >> 3, seg = (p & 7) * 8;
            async_cp16(Kh + (size_t)(kv0 + row) * D_K + seg, &kbuf[buf][row][seg]);
            async_cp16(Vh + (size_t)row * S_LEN + kv0 + seg, &vbuf[buf][row][seg]);
        }
    };

    // Q A-fragments for dk=64 (two K-steps of 32), loaded once
    v16bf aq[2];
    {
        const int m = q0 + l15;
#pragma unroll
        for (int ks = 0; ks < 2; ++ks) {
            const __bf16* qr = Qh + (size_t)m * D_K + ks * 32 + kbA;
            const v8bf lo = *(const v8bf*)qr;
            const v8bf hi = *(const v8bf*)(qr + 8);
            v16bf a;
#pragma unroll
            for (int e = 0; e < 8; ++e) { a[e] = lo[e]; a[8 + e] = hi[e]; }
            aq[ks] = a;
        }
    }

    v8f ctx[4];
#pragma unroll
    for (int t = 0; t < 4; ++t)
#pragma unroll
        for (int r = 0; r < 8; ++r) ctx[t][r] = 0.0f;
    float rowmax[8], rowsum[8];
#pragma unroll
    for (int r = 0; r < 8; ++r) { rowmax[r] = -1e30f; rowsum[r] = 0.0f; }

    const int NCH = S_LEN / 64;                    // 16 chunks
    stage(0, 0);

    for (int c = 0; c < NCH; ++c) {
        const int cur = c & 1;
        const int kv  = c * 64;

        if (c + 1 < NCH) {
            stage(c + 1, 1 - cur);
            asm volatile("s_wait_asynccnt 4" ::: "memory");  // chunk c done, c+1 in flight
        } else {
            asm volatile("s_wait_asynccnt 0" ::: "memory");
        }
        __syncthreads();                            // chunk c visible to all waves

        // ---- scores 16x64: 4 N-tiles x 2 K-steps = 8 WMMA (K from LDS) ----
        v8f sc[4];
#pragma unroll
        for (int tt = 0; tt < 4; ++tt) {
#pragma unroll
            for (int r = 0; r < 8; ++r) sc[tt][r] = 0.0f;
            const int n = tt * 16 + l15;           // local key row
#pragma unroll
            for (int ks = 0; ks < 2; ++ks) {
                const v16bf bk = *(const v16bf*)&kbuf[cur][n][ks * 32 + kbB];
                sc[tt] = wmma_bf16(aq[ks], bk, sc[tt]);
            }
        }

        // ---- scale + mask (mask is per key column) ----
        int mv[4];
#pragma unroll
        for (int tt = 0; tt < 4; ++tt) mv[tt] = mrow[kv + tt * 16 + l15];
#pragma unroll
        for (int tt = 0; tt < 4; ++tt)
#pragma unroll
            for (int r = 0; r < 8; ++r)
                sc[tt][r] = (mv[tt] == 0) ? -10000.0f : sc[tt][r] * 0.125f;

        // ---- online softmax: reductions stay within each 16-lane half ----
        float corr[8];
#pragma unroll
        for (int r = 0; r < 8; ++r) {
            float mloc = fmaxf(fmaxf(sc[0][r], sc[1][r]), fmaxf(sc[2][r], sc[3][r]));
#pragma unroll
            for (int off = 1; off < 16; off <<= 1)
                mloc = fmaxf(mloc, __shfl_xor(mloc, off, 32));
            const float newm = fmaxf(rowmax[r], mloc);
            corr[r]   = __expf(rowmax[r] - newm);
            rowmax[r] = newm;
        }
#pragma unroll
        for (int r = 0; r < 8; ++r) {
            float ps = 0.0f;
#pragma unroll
            for (int tt = 0; tt < 4; ++tt) {
                sc[tt][r] = __expf(sc[tt][r] - rowmax[r]);
                ps += sc[tt][r];
            }
#pragma unroll
            for (int off = 1; off < 16; off <<= 1)
                ps += __shfl_xor(ps, off, 32);
            rowsum[r] = rowsum[r] * corr[r] + ps;
        }
#pragma unroll
        for (int t = 0; t < 4; ++t)
#pragma unroll
            for (int r = 0; r < 8; ++r) ctx[t][r] *= corr[r];

        // ---- P: C-layout -> A-fragment layout via wave-private LDS tile ----
#pragma unroll
        for (int r = 0; r < 8; ++r)
#pragma unroll
            for (int tt = 0; tt < 4; ++tt)
                pbuf[wave][r + half * 8][tt * 16 + l15] = (__bf16)sc[tt][r];
        asm volatile("s_wait_dscnt 0" ::: "memory");
        v16bf pa[2];
#pragma unroll
        for (int ks = 0; ks < 2; ++ks) {
#pragma unroll
            for (int e = 0; e < 8; ++e) {
                pa[ks][e]     = pbuf[wave][l15][ks * 32 + kbA + e];
                pa[ks][8 + e] = pbuf[wave][l15][ks * 32 + kbA + 8 + e];
            }
        }

        // ---- ctx += P(16x64) x V(64x64): 8 WMMA (V from LDS) ----
#pragma unroll
        for (int t = 0; t < 4; ++t) {
            const int n = t * 16 + l15;            // dk column = Vt row
#pragma unroll
            for (int ks = 0; ks < 2; ++ks) {
                const v16bf bv = *(const v16bf*)&vbuf[cur][n][ks * 32 + kbB];
                ctx[t] = wmma_bf16(pa[ks], bv, ctx[t]);
            }
        }

        __syncthreads();                            // all waves done with buf[cur]
    }

    // ---- normalize and write ctx as bf16 [B,S,H*dk] ----
    float inv[8];
#pragma unroll
    for (int r = 0; r < 8; ++r) inv[r] = 1.0f / rowsum[r];
#pragma unroll
    for (int t = 0; t < 4; ++t) {
        const int d = h * D_K + t * 16 + l15;
#pragma unroll
        for (int r = 0; r < 8; ++r) {
            const int srow = q0 + r + half * 8;
            Cout[((size_t)b * S_LEN + srow) * MODEL_DIM + d] = (__bf16)(ctx[t][r] * inv[r]);
        }
    }
}

extern "C" void kernel_launch(void* const* d_in, const int* in_sizes, int n_in,
                              void* d_out, int out_size, void* d_ws, size_t ws_size,
                              hipStream_t stream) {
    const float* query = (const float*)d_in[0];
    const float* key_  = (const float*)d_in[1];
    const float* value = (const float*)d_in[2];
    const int*   mask  = (const int*)d_in[3];
    const float* Wq = (const float*)d_in[4];
    const float* bq = (const float*)d_in[5];
    const float* Wk = (const float*)d_in[6];
    const float* bk = (const float*)d_in[7];
    const float* Wv = (const float*)d_in[8];
    const float* bv = (const float*)d_in[9];
    const float* Wo = (const float*)d_in[10];
    const float* bo = (const float*)d_in[11];
    float* out = (float*)d_out;

    // Workspace (bf16 elements): 5 x 8M activations + 4 x 1M weights = 88 MB
    const size_t actN = (size_t)BATCH * S_LEN * MODEL_DIM;   // 8M
    const size_t wN   = (size_t)MODEL_DIM * MODEL_DIM;       // 1M
    __bf16* Qbf = (__bf16*)d_ws;
    __bf16* Kbf = Qbf + actN;
    __bf16* Vtb = Kbf + actN;
    __bf16* Cbf = Vtb + actN;
    __bf16* Xbf = Cbf + actN;
    __bf16* WqB = Xbf + actN;
    __bf16* WkB = WqB + wN;
    __bf16* WvB = WkB + wN;
    __bf16* WoB = WvB + wN;

    const dim3 blk(128);
    const int M = BATCH * S_LEN;                                   // 8192
    const int projBlocks = (M / 32) * (MODEL_DIM / 64) / 4;        // 1024
    const int attnBlocks = BATCH * NUM_HEADS * (S_LEN / 128);      // 1024
    const int cvtW = (int)(wN / 8 / 256);                          // 512
    const int cvtX = (int)(actN / 8 / 256);                        // 4096

    cvt_bf16<<<cvtW, 256, 0, stream>>>(Wq, WqB, (int)wN);
    cvt_bf16<<<cvtW, 256, 0, stream>>>(Wk, WkB, (int)wN);
    cvt_bf16<<<cvtW, 256, 0, stream>>>(Wv, WvB, (int)wN);
    cvt_bf16<<<cvtW, 256, 0, stream>>>(Wo, WoB, (int)wN);

    cvt_bf16<<<cvtX, 256, 0, stream>>>(query, Xbf, (int)actN);
    proj_wmma<<<projBlocks, blk, 0, stream>>>(Xbf, WqB, bq, nullptr, Qbf, 1);
    cvt_bf16<<<cvtX, 256, 0, stream>>>(key_, Xbf, (int)actN);
    proj_wmma<<<projBlocks, blk, 0, stream>>>(Xbf, WkB, bk, nullptr, Kbf, 1);
    cvt_bf16<<<cvtX, 256, 0, stream>>>(value, Xbf, (int)actN);
    proj_wmma<<<projBlocks, blk, 0, stream>>>(Xbf, WvB, bv, nullptr, Vtb, 2);

    attn_wmma<<<attnBlocks, 256, 0, stream>>>(Qbf, Kbf, Vtb, mask, Cbf);
    proj_wmma<<<projBlocks, blk, 0, stream>>>(Cbf, WoB, bo, out, nullptr, 0);
}
